// SwinAttention_43550968381942
// MI455X (gfx1250) — matrix operand order repaired and checked by reading, not compiled, not run on hardware
//
#include <hip/hip_runtime.h>

typedef __attribute__((ext_vector_type(16))) _Float16 v16h;
typedef __attribute__((ext_vector_type(8)))  _Float16 v8h;
typedef __attribute__((ext_vector_type(8)))  float    v8f;

__device__ __forceinline__ v16h load_a16(const _Float16* p0, const _Float16* p1) {
    union { v16h v; v8h h[2]; } u;
    u.h[0] = *(const v8h*)p0;
    u.h[1] = *(const v8h*)p1;
    return u.v;
}

// Region id of window-token t for the Swin shift mask (H=112, w=7, s=3).
__device__ __forceinline__ int region_of(int t, int lastY, int lastX) {
    int ty = t / 7, tx = t - ty * 7;
    int rr = lastY ? (ty < 4 ? 1 : 2) : 0;
    int rc = lastX ? (tx < 4 ? 1 : 2) : 0;
    return rr * 3 + rc;
}

// Convert the three 128x128 fp32 weight matrices to f16 in workspace.
__global__ void swin_convert_weights(const float* __restrict__ Wq,
                                     const float* __restrict__ Wk,
                                     const float* __restrict__ Wv,
                                     _Float16* __restrict__ wf) {
    int i = blockIdx.x * 256 + threadIdx.x;
    if (i < 16384)      wf[i] = (_Float16)Wq[i];
    else if (i < 32768) wf[i] = (_Float16)Wk[i - 16384];
    else if (i < 49152) wf[i] = (_Float16)Wv[i - 32768];
}

// One workgroup (8 wave32) per 7x7 window: fused QKV proj + masked attention.
__global__ __launch_bounds__(256)
void swin_window_attn(const float* __restrict__ x,
                      const _Float16* __restrict__ wf,
                      const float* __restrict__ bq,
                      const float* __restrict__ bk,
                      const float* __restrict__ bv,
                      float* __restrict__ out) {
    __shared__ __align__(32) unsigned char smem[65536];
    _Float16* xh = (_Float16*)smem;            // 64x128 f16 x-tile (later reused as scores/attn)
    _Float16* qh = (_Float16*)(smem + 16384);  // 64x128 f16 q (flat head slices)
    _Float16* kh = (_Float16*)(smem + 32768);  // 64x128 f16 k
    _Float16* vt = (_Float16*)(smem + 49152);  // per-head transposed v: [4][32][64] f16
    float*    sc = (float*)smem;               // 64x64 f32 scores (per head)

    const int tid  = threadIdx.x;
    const int lane = tid & 31;
    const int wave = tid >> 5;
    const int lrow = lane & 15;   // column/row-within-halfwave
    const int hiH  = lane >> 4;   // 0 or 1: which half of the wave

    const int wid = blockIdx.x;        // 0..8191
    const int b   = wid >> 8;
    const int wI  = wid & 255;
    const int wy  = wI >> 4;
    const int wx  = wI & 15;
    const int lastY = (wy == 15);
    const int lastX = (wx == 15);

    // ---- Stage rolled x window tile to LDS f16 (rows 49..63 zero); zero vT.
    for (int idx = tid; idx < 8192; idx += 256) {
        int row = idx >> 7, col = idx & 127;
        float v = 0.f;
        if (row < 49) {
            int t7 = row / 7;
            int r = wy * 7 + t7, c = wx * 7 + (row - t7 * 7);
            int rs = r + 3; if (rs >= 112) rs -= 112;
            int cs = c + 3; if (cs >= 112) cs -= 112;
            v = x[(((size_t)b * 112 + rs) * 112 + cs) * 128 + col];
        }
        xh[idx] = (_Float16)v;
        vt[idx] = (_Float16)0.f;
    }
    __syncthreads();

    // ---- QKV projection: out = x_tile @ W^T + b.  96 tiles of 16x16, K=128.
    for (int T = wave; T < 96; T += 8) {
        int mat = T >> 5, rem = T & 31, mt = rem >> 3, nt = rem & 7;
        const _Float16* W = wf + mat * 16384;               // W[n][k], k contiguous
        const float* bias = (mat == 0) ? bq : ((mat == 1) ? bk : bv);
        int col = nt * 16 + lrow;
        v8f acc = {};
        for (int kt = 0; kt < 4; ++kt) {
            const _Float16* ap = xh + (mt * 16 + lrow) * 128 + kt * 32 + hiH * 8;
            v16h A = load_a16(ap, ap + 16);
            const _Float16* bp = W + col * 128 + kt * 32 + hiH * 8;
            v16h Bf = load_a16(bp, bp + 16);
            acc = __builtin_amdgcn_wmma_f32_16x16x32_f16(false, A, false, Bf,
                                                         (short)0, acc, false, false);
        }
        float bb = bias[col];
        if (mat < 2) {
            _Float16* dst = (mat == 0) ? qh : kh;
            for (int i = 0; i < 8; ++i) {
                int row = mt * 16 + hiH * 8 + i;
                dst[row * 128 + col] = (_Float16)(acc[i] + bb);
            }
        } else {
            // v: scatter into per-head transposed layout via the reference's
            // raw (nh, ws2, hd) reshape: flat f -> h = f/1568, j = (f%1568)/32, d = f%32.
            for (int i = 0; i < 8; ++i) {
                unsigned f = (unsigned)(mt * 16 + hiH * 8 + i) * 128u + (unsigned)col;
                if (f < 6272u) {
                    unsigned h  = f / 1568u;
                    unsigned r2 = f - h * 1568u;
                    unsigned j  = r2 >> 5, d = r2 & 31u;
                    vt[(h * 32 + d) * 64 + j] = (_Float16)(acc[i] + bb);
                }
            }
        }
    }

    const float scale = 0.1767766952966369f; // 1/sqrt(32)

    for (int h = 0; h < 4; ++h) {
        __syncthreads(); // proj->attn (h==0); scores-buffer reuse otherwise

        // ---- scores = q_h @ k_h^T * scale + mask.  16 tiles (64x64), K=32.
        for (int T = wave; T < 16; T += 8) {
            int mt = T >> 2, nt = T & 3;
            const _Float16* ap = qh + h * 1568 + (mt * 16 + lrow) * 32 + hiH * 8;
            v16h A = load_a16(ap, ap + 16);
            int ncol = nt * 16 + lrow;
            const _Float16* bp = kh + h * 1568 + ncol * 32 + hiH * 8;
            v16h Bf = load_a16(bp, bp + 16);
            v8f acc = {};
            acc = __builtin_amdgcn_wmma_f32_16x16x32_f16(false, A, false, Bf,
                                                         (short)0, acc, false, false);
            int rn = region_of(ncol, lastY, lastX);
            for (int i = 0; i < 8; ++i) {
                int m  = mt * 16 + hiH * 8 + i;
                int rm = region_of(m, lastY, lastX);
                sc[m * 64 + ncol] = acc[i] * scale + ((rm == rn) ? 0.f : -100.f);
            }
        }
        __syncthreads();

        // ---- Lane-parallel row softmax (wave32): each wave owns 8 rows; each
        // lane covers 2 of 64 cols; max/sum via shfl_xor butterflies. The f16
        // attn row is written in place over the f32 scores row — safe because
        // all lane reads of the row issue before any lane's store (lockstep).
        for (int row = wave; row < 64; row += 8) {
            float* srow = sc + row * 64;
            _Float16* arow = (_Float16*)srow;
            if (row < 49) {
                float c0 = (lane < 49)      ? srow[lane]      : -1e30f;
                float c1 = (lane + 32 < 49) ? srow[lane + 32] : -1e30f;
                float mx = fmaxf(c0, c1);
                for (int off = 16; off > 0; off >>= 1)
                    mx = fmaxf(mx, __shfl_xor(mx, off, 32));
                float e0 = (lane < 49)      ? __expf(c0 - mx) : 0.f;
                float e1 = (lane + 32 < 49) ? __expf(c1 - mx) : 0.f;
                float sum = e0 + e1;
                for (int off = 16; off > 0; off >>= 1)
                    sum += __shfl_xor(sum, off, 32);
                float inv = 1.f / sum;
                arow[lane]      = (_Float16)(e0 * inv);
                arow[lane + 32] = (_Float16)(e1 * inv);
            } else {            // zero pad rows 49..63 so AV A-operand is clean
                arow[lane]      = (_Float16)0.f;
                arow[lane + 32] = (_Float16)0.f;
            }
        }
        __syncthreads();

        // ---- out_h = attn @ v_h.  8 tiles (64x32), K=64 -> one tile per wave.
        {
            int mt = wave >> 1, ntd = wave & 1;
            v8f acc = {};
            for (int kt = 0; kt < 2; ++kt) {
                const _Float16* ap = (const _Float16*)(sc + (mt * 16 + lrow) * 64)
                                     + kt * 32 + hiH * 8;
                v16h A = load_a16(ap, ap + 16);
                int d = ntd * 16 + lrow;
                const _Float16* bp = vt + (h * 32 + d) * 64 + kt * 32 + hiH * 8;
                v16h Bf = load_a16(bp, bp + 16);
                acc = __builtin_amdgcn_wmma_f32_16x16x32_f16(false, A, false, Bf,
                                                             (short)0, acc, false, false);
            }
            for (int i = 0; i < 8; ++i) {
                int t = mt * 16 + hiH * 8 + i;
                if (t < 49) {
                    int t7 = t / 7;
                    int r = wy * 7 + t7, c = wx * 7 + (t - t7 * 7);
                    int rs = r + 3; if (rs >= 112) rs -= 112;
                    int cs = c + 3; if (cs >= 112) cs -= 112;
                    out[(((size_t)b * 112 + rs) * 112 + cs) * 128
                        + h * 32 + ntd * 16 + lrow] = acc[i];
                }
            }
        }
    }
}

extern "C" void kernel_launch(void* const* d_in, const int* in_sizes, int n_in,
                              void* d_out, int out_size, void* d_ws, size_t ws_size,
                              hipStream_t stream) {
    const float* x  = (const float*)d_in[0];
    const float* Wq = (const float*)d_in[1];
    const float* bq = (const float*)d_in[2];
    const float* Wk = (const float*)d_in[3];
    const float* bk = (const float*)d_in[4];
    const float* Wv = (const float*)d_in[5];
    const float* bv = (const float*)d_in[6];
    float* out = (float*)d_out;
    _Float16* wf = (_Float16*)d_ws;   // 3*128*128 f16 = 98304 bytes of scratch
    (void)in_sizes; (void)n_in; (void)out_size; (void)ws_size;

    swin_convert_weights<<<192, 256, 0, stream>>>(Wq, Wk, Wv, wf);
    swin_window_attn<<<8192, 256, 0, stream>>>(x, wf, bq, bk, bv, out);
}
